// Context_atten_42305427865723
// MI455X (gfx1250) — matrix operand order repaired and checked by reading, not compiled, not run on hardware
//
#include <hip/hip_runtime.h>
#include <hip/hip_bf16.h>

// ---------------------------------------------------------------------------
// Context attention forward for MI455X (gfx1250, wave32, WMMA).
//   Mt:(T,H) f32, hc:(1,H), Wm:(H,H), V:(H,1), W1:(H,H)
//   At = tanh(Mt@Wm + hc@W1^T) @ V ; alphat = log_softmax(At) ; ct = alphat@Mt
// Heavy GEMM done with v_wmma_f32_16x16x32_bf16 using a bf16 hi/lo split
// (A*B ~= Ahi*Bhi + Ahi*Blo + Alo*Bhi) for ~f32 accuracy at bf16 rates.
// M-dimension register blocking: each wave owns 32 rows (two 16-row A tiles)
// so each LDS B-fragment load pair feeds 6 WMMAs -> inner loop is WMMA-bound,
// not ds_load-bound.
// ---------------------------------------------------------------------------

#define T_CTX 100000
#define H_DIM 256
#define NPANELS (T_CTX / 32)       // 3125 row panels of 32
#define WAVES_PER_BLOCK 8

typedef __attribute__((ext_vector_type(16))) __bf16 v16bf;
typedef __attribute__((ext_vector_type(8)))  float  v8f;

// ---------------------------------------------------------------------------
// Kernel 0: proj[n] = sum_h hc[h] * W1[n*H + h]   (hc @ W1^T)
// ---------------------------------------------------------------------------
__global__ __launch_bounds__(256) void proj_kernel(
    const float* __restrict__ hc, const float* __restrict__ W1,
    float* __restrict__ proj)
{
    __shared__ float sh[H_DIM];
    const int tid = threadIdx.x;
    sh[tid] = hc[tid];
    __syncthreads();
    const float* row = W1 + (size_t)tid * H_DIM;
    float acc = 0.f;
#pragma unroll 8
    for (int h = 0; h < H_DIM; ++h) acc += sh[h] * row[h];
    proj[tid] = acc;
}

// ---------------------------------------------------------------------------
// Kernel 1: per 32-row panel, scores = A@Wm + proj, At = tanh(scores)@V.
// Wm staged per-workgroup into LDS in WMMA B-fragment order, split hi/lo bf16
// (2 * 128KB LDS). Each wave: 2 A tiles x 16 N-tiles x 8 K-steps x 3 WMMAs;
// tanh + V-dot fused in-register, only At (T floats) hits memory.
//
// Fragment layouts per CDNA5 ISA 7.12.2:
//   A (16x32 bf16): lane m = lane&15; elem e=2v+h -> k = (v>=4?16:0)
//                   + (lane>=16?8:0) + 2*(v&3) + h
//   B (32x16 bf16): lane n = lane&15; k = (e>=8?16:0) + (lane>=16?8:0) + (e&7)  (e=0..15)
//   C/D (16x16 f32): col = lane&15, row = (lane>>4)*8 + r
// ---------------------------------------------------------------------------
__global__ __launch_bounds__(256) void scores_kernel(
    const float* __restrict__ Mt, const float* __restrict__ Wm,
    const float* __restrict__ proj, const float* __restrict__ V,
    float* __restrict__ At)
{
    extern __shared__ __bf16 lds[];            // [hi | lo], each 65536 bf16
    __bf16* ldsHi = lds;
    __bf16* ldsLo = lds + H_DIM * H_DIM;

    const int tid = threadIdx.x;

    // ---- stage Wm -> LDS, pre-split + pre-swizzled into B-fragment order ---
    for (int idx = tid; idx < H_DIM * H_DIM; idx += 256) {
        const int k = idx >> 8, n = idx & 255;         // Wm row-major (k,n)
        const float w = Wm[idx];
        const __bf16 hi = (__bf16)w;
        const __bf16 lo = (__bf16)(w - (float)hi);
        const int nt = n >> 4, nl = n & 15, kc = k >> 5, kl = k & 31;
        const int blane = (((kl >> 3) & 1) << 4) | nl; // fragment lane
        const int e     = (kl & 7) | ((kl >> 4) << 3); // element within lane
        const int off = (((nt * 8 + kc) * 32) + blane) * 16 + e;
        ldsHi[off] = hi;
        ldsLo[off] = lo;
    }
    __syncthreads();

    const int wave  = tid >> 5;
    const int lane  = tid & 31;
    const int panel = blockIdx.x * WAVES_PER_BLOCK + wave;
    if (panel >= NPANELS) return;                      // uniform per wave

    const int lanehi = lane >> 4;
    const int nl     = lane & 15;
    const size_t r0  = (size_t)panel * 32;             // rows r0..r0+31

    // ---- load & split two A tiles (2 x 16 rows x 256 K) into hi/lo frags --
    v16bf ahi0[8], alo0[8], ahi1[8], alo1[8];
    const float* arow0 = Mt + (r0 + (size_t)nl) * H_DIM;        // tile 0 row
    const float* arow1 = arow0 + (size_t)16 * H_DIM;            // tile 1 row
#pragma unroll
    for (int kc = 0; kc < 8; ++kc) {
#pragma unroll
        for (int v = 0; v < 8; ++v) {
            const int k0 = kc * 32 + ((v >= 4) ? 16 : 0) + lanehi * 8 + 2 * (v & 3);
            const float2 w0 = *(const float2*)(arow0 + k0);
            const float2 w1 = *(const float2*)(arow1 + k0);
            const __bf16 a0 = (__bf16)w0.x, a1 = (__bf16)w0.y;
            const __bf16 b0 = (__bf16)w1.x, b1 = (__bf16)w1.y;
            ahi0[kc][2 * v]     = a0;
            ahi0[kc][2 * v + 1] = a1;
            alo0[kc][2 * v]     = (__bf16)(w0.x - (float)a0);
            alo0[kc][2 * v + 1] = (__bf16)(w0.y - (float)a1);
            ahi1[kc][2 * v]     = b0;
            ahi1[kc][2 * v + 1] = b1;
            alo1[kc][2 * v]     = (__bf16)(w1.x - (float)b0);
            alo1[kc][2 * v + 1] = (__bf16)(w1.y - (float)b1);
        }
    }

    float atp0[8], atp1[8];
#pragma unroll
    for (int r = 0; r < 8; ++r) { atp0[r] = 0.f; atp1[r] = 0.f; }

    for (int nt = 0; nt < 16; ++nt) {
        v8f acc0 = {};
        v8f acc1 = {};
#pragma unroll
        for (int kc = 0; kc < 8; ++kc) {
            const int fb = (((nt * 8 + kc) * 32) + lane) * 16;
            const v16bf bhi = *(const v16bf*)(ldsHi + fb);   // 32B contiguous
            const v16bf blo = *(const v16bf*)(ldsLo + fb);
            acc0 = __builtin_amdgcn_wmma_f32_16x16x32_bf16(
                       false, ahi0[kc], false, bhi, (short)0, acc0, false, false);
            acc1 = __builtin_amdgcn_wmma_f32_16x16x32_bf16(
                       false, ahi1[kc], false, bhi, (short)0, acc1, false, false);
            acc0 = __builtin_amdgcn_wmma_f32_16x16x32_bf16(
                       false, ahi0[kc], false, blo, (short)0, acc0, false, false);
            acc1 = __builtin_amdgcn_wmma_f32_16x16x32_bf16(
                       false, ahi1[kc], false, blo, (short)0, acc1, false, false);
            acc0 = __builtin_amdgcn_wmma_f32_16x16x32_bf16(
                       false, alo0[kc], false, bhi, (short)0, acc0, false, false);
            acc1 = __builtin_amdgcn_wmma_f32_16x16x32_bf16(
                       false, alo1[kc], false, bhi, (short)0, acc1, false, false);
        }
        const int   n  = nt * 16 + nl;
        const float pn = proj[n];
        const float vn = V[n];
#pragma unroll
        for (int r = 0; r < 8; ++r) {
            atp0[r] += tanhf(acc0[r] + pn) * vn;   // row r0    + lanehi*8+r
            atp1[r] += tanhf(acc1[r] + pn) * vn;   // row r0+16 + lanehi*8+r
        }
    }

    // ---- reduce over the 16 lanes sharing each output row ------------------
#pragma unroll
    for (int r = 0; r < 8; ++r) {
        float v0 = atp0[r];
        v0 += __shfl_xor(v0, 1, 32);
        v0 += __shfl_xor(v0, 2, 32);
        v0 += __shfl_xor(v0, 4, 32);
        v0 += __shfl_xor(v0, 8, 32);
        atp0[r] = v0;
        float v1 = atp1[r];
        v1 += __shfl_xor(v1, 1, 32);
        v1 += __shfl_xor(v1, 2, 32);
        v1 += __shfl_xor(v1, 4, 32);
        v1 += __shfl_xor(v1, 8, 32);
        atp1[r] = v1;
    }
    if (nl < 8) {
        float v0 = atp0[0];
        float v1 = atp1[0];
#pragma unroll
        for (int r = 1; r < 8; ++r) {
            if (nl == r) { v0 = atp0[r]; v1 = atp1[r]; }
        }
        At[r0 + (size_t)lanehi * 8 + nl]      = v0;
        At[r0 + 16 + (size_t)lanehi * 8 + nl] = v1;
    }
}

// ---------------------------------------------------------------------------
// Kernel 2: single-block online logsumexp over At; also zero ct accumulators.
// ---------------------------------------------------------------------------
__global__ __launch_bounds__(1024) void lse_kernel(
    const float* __restrict__ At, float* __restrict__ lse,
    float* __restrict__ ct_zero)
{
    __shared__ float sm[1024];
    __shared__ float ss[1024];
    const int tid = threadIdx.x;

    float m = -INFINITY, s = 0.f;
    for (int t = tid; t < T_CTX; t += 1024) {
        const float x = At[t];
        if (x > m) { s = s * expf(m - x) + 1.f; m = x; }
        else       { s += expf(x - m); }
    }
    sm[tid] = m; ss[tid] = s;
    __syncthreads();
    for (int off = 512; off > 0; off >>= 1) {
        if (tid < off) {
            const float m1 = sm[tid], s1 = ss[tid];
            const float m2 = sm[tid + off], s2 = ss[tid + off];
            const float mm = fmaxf(m1, m2);
            sm[tid] = mm;
            ss[tid] = s1 * expf(m1 - mm) + s2 * expf(m2 - mm);
        }
        __syncthreads();
    }
    if (tid == 0) lse[0] = sm[0] + logf(ss[0]);
    if (tid < H_DIM) ct_zero[tid] = 0.f;    // deterministic re-zero each call
}

// ---------------------------------------------------------------------------
// Kernel 3: alphat[t] = At[t] - lse (written to out[0..T)); ct[n] += sum_t
// alphat[t]*Mt[t,n] via per-block partials + atomicAdd (out[T..T+H)).
// ---------------------------------------------------------------------------
#define CT_CHUNK 512
__global__ __launch_bounds__(256) void ct_kernel(
    const float* __restrict__ Mt, const float* __restrict__ At,
    const float* __restrict__ lse_p, float* __restrict__ out)
{
    __shared__ float sa[CT_CHUNK];
    const int   tid  = threadIdx.x;
    const int   base = blockIdx.x * CT_CHUNK;
    const int   nvalid = min(CT_CHUNK, T_CTX - base);
    const float lse  = lse_p[0];

    for (int j = tid; j < CT_CHUNK; j += 256) {
        const int t = base + j;
        if (t < T_CTX) {
            const float a = At[t] - lse;
            sa[j]  = a;
            out[t] = a;                       // alphat output
        }
    }
    __syncthreads();

    float acc = 0.f;
    int i = 0;
    for (; i + 4 <= nvalid; i += 4) {         // coalesced 1KB row reads
        const size_t rb = (size_t)(base + i) * H_DIM + tid;
        acc += sa[i + 0] * Mt[rb + 0 * H_DIM];
        acc += sa[i + 1] * Mt[rb + 1 * H_DIM];
        acc += sa[i + 2] * Mt[rb + 2 * H_DIM];
        acc += sa[i + 3] * Mt[rb + 3 * H_DIM];
    }
    for (; i < nvalid; ++i)
        acc += sa[i] * Mt[(size_t)(base + i) * H_DIM + tid];

    atomicAdd(&out[T_CTX + tid], acc);
}

// ---------------------------------------------------------------------------
extern "C" void kernel_launch(void* const* d_in, const int* in_sizes, int n_in,
                              void* d_out, int out_size, void* d_ws, size_t ws_size,
                              hipStream_t stream)
{
    const float* Mt = (const float*)d_in[0];   // (T,H)
    const float* hc = (const float*)d_in[1];   // (1,H)
    const float* Wm = (const float*)d_in[2];   // (H,H)
    const float* V  = (const float*)d_in[3];   // (H,1)
    const float* W1 = (const float*)d_in[4];   // (H,H)
    float* out = (float*)d_out;                // [alphat(T) | ct(H)]
    float* ws  = (float*)d_ws;

    float* At   = ws;                 // T floats
    float* lse  = ws + T_CTX;         // 1 float
    float* proj = ws + T_CTX + 1;     // H floats

    proj_kernel<<<1, 256, 0, stream>>>(hc, W1, proj);

    const int gblocks = (NPANELS + WAVES_PER_BLOCK - 1) / WAVES_PER_BLOCK;
    const size_t ldsBytes = (size_t)2 * H_DIM * H_DIM * 2;   // hi+lo bf16 = 256KB
    scores_kernel<<<gblocks, 256, ldsBytes, stream>>>(Mt, Wm, proj, V, At);

    lse_kernel<<<1, 1024, 0, stream>>>(At, lse, out + T_CTX);

    const int cblocks = (T_CTX + CT_CHUNK - 1) / CT_CHUNK;
    ct_kernel<<<cblocks, 256, 0, stream>>>(Mt, At, lse, out);
}
